// Loss_42941083025865
// MI455X (gfx1250) — matrix operand order repaired and checked by reading, not compiled, not run on hardware
//
#include <hip/hip_runtime.h>

#define Bc 256
#define Lc 1024
#define Kc 64
#define ROWS 72   // LDS row stride in halves: 144B rows -> conflict-free b32/b128 reads

typedef __attribute__((ext_vector_type(16))) _Float16 v16h;
typedef __attribute__((ext_vector_type(8)))  float    v8f;

union HFrag { v16h v; _Float16 h[16]; unsigned u[8]; };

// Max over each 16-lane row (lanes 0-15 and 16-31 reduce independently) using
// DPP16 row ops only -> pure VALU, no LDS-pipe round trips on the critical path.
__device__ __forceinline__ float rowmax16(float x) {
  int xi, y;
  xi = __float_as_int(x);
  y  = __builtin_amdgcn_update_dpp(xi, xi, 0xB1, 0xF, 0xF, false);  // quad_perm [1,0,3,2]
  x  = fmaxf(x, __int_as_float(y));
  xi = __float_as_int(x);
  y  = __builtin_amdgcn_update_dpp(xi, xi, 0x4E, 0xF, 0xF, false);  // quad_perm [2,3,0,1]
  x  = fmaxf(x, __int_as_float(y));
  xi = __float_as_int(x);
  y  = __builtin_amdgcn_update_dpp(xi, xi, 0x141, 0xF, 0xF, false); // row_half_mirror
  x  = fmaxf(x, __int_as_float(y));
  xi = __float_as_int(x);
  y  = __builtin_amdgcn_update_dpp(xi, xi, 0x140, 0xF, 0xF, false); // row_mirror
  x  = fmaxf(x, __int_as_float(y));
  return x;
}

// ---------------------------------------------------------------------------
// Forward algorithm (denominator). One wave = 16 batches, barrier-free.
// State: (ntot[b] : int, A[b][i] : f16 in LDS) with alpha = ntot*ln2 + log A,
// A normalized into (0.5, 1] by exact power-of-two scaling.
// Step: S = A x exp(T) via v_wmma_f32_16x16x32_f16 (2 per 16x16 N-tile, 4
// tiles), u = S*exp(em), n = exponent(max u), ntot += n, A' = u * 2^-n.
// PREEXP=1: em already exponentiated into workspace by exp_em_kernel.
// ---------------------------------------------------------------------------
template <int PREEXP>
__global__ __launch_bounds__(32)
void crf_fwd_kernel(const float* __restrict__ emx,   // em or exp(em)
                    const float* __restrict__ em,    // raw em (for t=0 init)
                    const float* __restrict__ startT,
                    const float* __restrict__ endT,
                    const float* __restrict__ trans,
                    const int*   __restrict__ mask,
                    float* __restrict__ logZ)
{
  __shared__ _Float16 expA[16 * ROWS];
  __shared__ int nS[16];

  const int lane = threadIdx.x;      // 0..31
  const int half = lane >> 4;        // 0/1
  const int l16  = lane & 15;
  const int b0   = blockIdx.x * 16;

  // ---- B fragments: exp(transitions) in WMMA B layout (loop-invariant) ----
  // B 32x16 f16: element e of lane L holds (K = 32*s + 16*(L/16) + e, N = L%16)
  v16h bfrag[4][2];
#pragma unroll
  for (int w = 0; w < 4; ++w)
#pragma unroll
    for (int s = 0; s < 2; ++s) {
      HFrag f;
#pragma unroll
      for (int e = 0; e < 16; ++e) {
        int i = 32 * s + 16 * half + e;
        int j = 16 * w + l16;
        f.h[e] = (_Float16)__expf(trans[i * Kc + j]);
      }
      bfrag[w][s] = f.v;
    }

  int ntot[8];   // exact pow2 normalizer exponent, b = r + 8*half

  // ---- t = 0 init: u0 = exp(start + em[:,0,:]) ----
  {
    float u[4][8];
#pragma unroll
    for (int w = 0; w < 4; ++w)
#pragma unroll
      for (int r = 0; r < 8; ++r) {
        int b = r + 8 * half;
        int j = 16 * w + l16;
        u[w][r] = __expf(startT[j] + em[(size_t)(b0 + b) * (Lc * Kc) + j]);
      }
    float sc[8];
#pragma unroll
    for (int r = 0; r < 8; ++r) {
      float mx = rowmax16(fmaxf(fmaxf(u[0][r], u[1][r]), fmaxf(u[2][r], u[3][r])));
      unsigned e = __float_as_uint(mx) >> 23;       // mx = f*2^n, f in [0.5,1)
      ntot[r] = (int)e - 126;
      sc[r]   = __uint_as_float((253u - e) << 23);  // exact 2^-n
    }
#pragma unroll
    for (int w = 0; w < 4; ++w)
#pragma unroll
      for (int r = 0; r < 8; ++r)
        expA[(r + 8 * half) * ROWS + 16 * w + l16] = (_Float16)(u[w][r] * sc[r]);
  }
  __syncthreads();   // single-wave block: NOP barrier, compiler fence only

  // ---- sequential forward recursion ----
  for (int t = 1; t < Lc; ++t) {
    // Emission + mask loads first: independent of the recurrence, overlap WMMA.
    float emv[4][8];
#pragma unroll
    for (int w = 0; w < 4; ++w)
#pragma unroll
      for (int r = 0; r < 8; ++r)
        emv[w][r] = emx[(size_t)(b0 + r + 8 * half) * (Lc * Kc) +
                        (size_t)t * Kc + 16 * w + l16];
    int mk[8];
#pragma unroll
    for (int r = 0; r < 8; ++r)
      mk[r] = mask[(b0 + r + 8 * half) * Lc + t];   // uniform over 16 lanes, WGP$

    // Prefetch next step's emission tile into near caches.
    int tn = (t + 1 < Lc) ? (t + 1) : t;
#pragma unroll
    for (int r = 0; r < 8; ++r)
      __builtin_prefetch(&emx[(size_t)(b0 + r + 8 * half) * (Lc * Kc) +
                              (size_t)tn * Kc + l16 * 4], 0, 3);

    // A fragments from LDS. contiguous k order (same as bfrag): pair p of lane L holds
    // K0 = 16*(L/16) + 2p, K0+1 for row M = L%16.
    HFrag a0, a1;
    const _Float16* rowp = &expA[l16 * ROWS];
#pragma unroll
    for (int p = 0; p < 8; ++p) {
      int K0 = 16 * half + 2 * p;
      __builtin_memcpy(&a0.u[p], rowp + K0, 4);        // merges to ds_load_b128
      __builtin_memcpy(&a1.u[p], rowp + K0 + 32, 4);
    }

    // S = A x exp(T): 4 independent tiles, 2 chained WMMAs each (K=64).
    v8f cw[4];
#pragma unroll
    for (int w = 0; w < 4; ++w) {
      v8f acc = {};
      acc = __builtin_amdgcn_wmma_f32_16x16x32_f16(false, a0.v, false, bfrag[w][0],
                                                   (short)0, acc, false, false);
      acc = __builtin_amdgcn_wmma_f32_16x16x32_f16(false, a1.v, false, bfrag[w][1],
                                                   (short)0, acc, false, false);
      cw[w] = acc;
    }

    // u = S * exp(em)
    float u[4][8];
#pragma unroll
    for (int w = 0; w < 4; ++w)
#pragma unroll
      for (int r = 0; r < 8; ++r)
        u[w][r] = PREEXP ? (cw[w][r] * emv[w][r])
                         : (cw[w][r] * __expf(emv[w][r]));

    // Per-batch max over all 64 states (4-tile max3 + DPP16 row reduction),
    // then exact pow2 renormalization.
    float sc[8];
#pragma unroll
    for (int r = 0; r < 8; ++r) {
      float mx = rowmax16(fmaxf(fmaxf(u[0][r], u[1][r]), fmaxf(u[2][r], u[3][r])));
      unsigned e = __float_as_uint(mx) >> 23;
      int n = (int)e - 126;
      ntot[r] += mk[r] ? n : 0;                 // exact integer accumulation
      sc[r]    = __uint_as_float((253u - e) << 23);
    }

    __syncthreads();
    // Masked step keeps old A: only store when mask==1.
#pragma unroll
    for (int w = 0; w < 4; ++w)
#pragma unroll
      for (int r = 0; r < 8; ++r)
        if (mk[r])
          expA[(r + 8 * half) * ROWS + 16 * w + l16] = (_Float16)(u[w][r] * sc[r]);
    __syncthreads();
  }

  // ---- logZ[b] = ntot[b]*ln2 + log( sum_j A[b][j] * exp(end[j]) ) ----
  if (l16 == 0) {
#pragma unroll
    for (int r = 0; r < 8; ++r) nS[r + 8 * half] = ntot[r];
  }
  __syncthreads();

  float s = 0.0f;
#pragma unroll 8
  for (int jj = 0; jj < 32; ++jj) {
    int j = 32 * half + jj;
    s += (float)expA[l16 * ROWS + j] * __expf(endT[j]);
  }
  s += __shfl_xor(s, 16, 32);
  if (half == 0)
    logZ[b0 + l16] =
        (float)((double)nS[l16] * 0.6931471805599453) + __logf(s);
}

// ---------------------------------------------------------------------------
// exp(emissions) precompute: fully parallel, bandwidth-bound.
// ---------------------------------------------------------------------------
__global__ __launch_bounds__(256)
void exp_em_kernel(const float4* __restrict__ in, float4* __restrict__ out)
{
  int i = blockIdx.x * 256 + threadIdx.x;   // grid sized exactly: n/4 threads
  float4 v = in[i];
  float4 o;
  o.x = __expf(v.x); o.y = __expf(v.y); o.z = __expf(v.z); o.w = __expf(v.w);
  out[i] = o;
}

// ---------------------------------------------------------------------------
// Numerator: gold-path score. One block per batch, parallel over t.
// ---------------------------------------------------------------------------
__global__ __launch_bounds__(256)
void crf_num_kernel(const float* __restrict__ em,
                    const float* __restrict__ startT,
                    const float* __restrict__ endT,
                    const float* __restrict__ trans,
                    const int*   __restrict__ mask,
                    const int*   __restrict__ tags,
                    float* __restrict__ num)
{
  __shared__ float sf[256];
  __shared__ int   si[256];
  int b   = blockIdx.x;
  int tid = threadIdx.x;
  const int*   tg = tags + b * Lc;
  const int*   mk = mask + b * Lc;
  const float* e  = em + (size_t)b * (Lc * Kc);

  float ps = 0.0f;
  int   pc = 0;
#pragma unroll
  for (int k = 0; k < Lc / 256; ++k) {
    int t = tid + 256 * k;
    int m = mk[t];
    pc += m;
    if (t >= 1 && m) {
      int cur = tg[t];
      ps += trans[tg[t - 1] * Kc + cur] + e[(size_t)t * Kc + cur];
    }
  }
  sf[tid] = ps; si[tid] = pc;
  __syncthreads();
  for (int s = 128; s > 0; s >>= 1) {
    if (tid < s) { sf[tid] += sf[tid + s]; si[tid] += si[tid + s]; }
    __syncthreads();
  }
  if (tid == 0) {
    int t0 = tg[0];
    num[b] = sf[0] + startT[t0] + e[t0] + endT[tg[si[0] - 1]];
  }
}

// ---------------------------------------------------------------------------
// Final: mean(logZ - numerator) -> scalar.
// ---------------------------------------------------------------------------
__global__ __launch_bounds__(256)
void crf_reduce_kernel(const float* __restrict__ logZ,
                       const float* __restrict__ num,
                       float* __restrict__ out)
{
  __shared__ float sm[256];
  int tid = threadIdx.x;
  sm[tid] = logZ[tid] - num[tid];
  __syncthreads();
  for (int s = 128; s > 0; s >>= 1) {
    if (tid < s) sm[tid] += sm[tid + s];
    __syncthreads();
  }
  if (tid == 0) out[0] = sm[0] * (1.0f / 256.0f);
}

extern "C" void kernel_launch(void* const* d_in, const int* in_sizes, int n_in,
                              void* d_out, int out_size, void* d_ws, size_t ws_size,
                              hipStream_t stream) {
  (void)in_sizes; (void)n_in; (void)out_size;
  const float* em     = (const float*)d_in[0];
  const float* startT = (const float*)d_in[1];
  const float* endT   = (const float*)d_in[2];
  const float* trans  = (const float*)d_in[3];
  const int*   mask   = (const int*)d_in[4];
  const int*   tags   = (const int*)d_in[5];

  float* num  = (float*)d_ws;                    // 256 floats
  float* logZ = num + Bc;                        // 256 floats
  float* expEm = (float*)((char*)d_ws + 2048);   // Bc*Lc*Kc floats (64 MB)
  float* out  = (float*)d_out;

  const size_t nEm = (size_t)Bc * Lc * Kc;
  const bool pre = ws_size >= 2048 + nEm * sizeof(float);

  crf_num_kernel<<<Bc, 256, 0, stream>>>(em, startT, endT, trans, mask, tags, num);

  if (pre) {
    exp_em_kernel<<<(int)(nEm / 4 / 256), 256, 0, stream>>>(
        (const float4*)em, (float4*)expEm);
    crf_fwd_kernel<1><<<Bc / 16, 32, 0, stream>>>(expEm, em, startT, endT, trans,
                                                  mask, logZ);
  } else {
    crf_fwd_kernel<0><<<Bc / 16, 32, 0, stream>>>(em, em, startT, endT, trans,
                                                  mask, logZ);
  }

  crf_reduce_kernel<<<1, 256, 0, stream>>>(logZ, num, out);
}
